// HeatCorner_1992864825646
// MI455X (gfx1250) — matrix operand order repaired and checked by reading, not compiled, not run on hardware
//
#include <hip/hip_runtime.h>
#include <hip/hip_bf16.h>

// ---------------------------------------------------------------------------
// HeatCorner forward for MI455X (gfx1250, wave32, WMMA).
// All GEMM-like work (linear layers + 3x3 convs as implicit GEMM) runs through
// v_wmma_f32_16x16x32_f16 with fp32 accumulation. Conv inputs are pre-packed
// channels-last f16 so every WMMA A/B fragment is fed by b128 loads; zero
// padding is a branch-free pointer select onto a zeroed scratch row. The
// linear-GEMM kernel stages its LDS tiles with GLOBAL_LOAD_ASYNC_TO_LDS_B128
// (ASYNCcnt) — the gfx1250 async global->LDS path.
// ---------------------------------------------------------------------------

#define USE_ASYNC_LDS 1

typedef __attribute__((ext_vector_type(16))) _Float16 v16h;
typedef __attribute__((ext_vector_type(8)))  _Float16 v8h;
typedef __attribute__((ext_vector_type(8)))  float    v8f;

#define D_MODEL 256
#define NHEAD   8
#define HDIM    32
#define FFDIM   1024
#define BATCH   4
#define LV_TOT  5440               // 4096+1024+256+64
#define RE      (BATCH * LV_TOT)   // 21760 encoder rows
#define LQ      4096
#define RD      (BATCH * LQ)       // 16384 decoder rows

__device__ __forceinline__ v16h cat8(v8h lo, v8h hi) {
  v16h r;
#pragma unroll
  for (int i = 0; i < 8; ++i) { r[i] = lo[i]; r[i + 8] = hi[i]; }
  return r;
}

// ---------------------------------------------------------------------------
// Weight prep: W (KxN f32) -> Wt (NxK f16).
// ---------------------------------------------------------------------------
__global__ void k_wt(const float* __restrict__ w, _Float16* __restrict__ wt,
                     int K, int N) {
  long idx = (long)blockIdx.x * blockDim.x + threadIdx.x;
  if (idx >= (long)K * N) return;
  int  n = (int)(idx % N);
  long k = idx / N;
  wt[(long)n * K + k] = (_Float16)w[idx];
}

// Conv weights OIHW f32 -> [ky*3+kx][Co][Ci] f16 (ci contiguous).
__global__ void k_cwt(const float* __restrict__ w, _Float16* __restrict__ wt,
                      int Co, int Ci) {
  long idx = (long)blockIdx.x * blockDim.x + threadIdx.x;
  long total = (long)Co * Ci * 9;
  if (idx >= total) return;
  long kk = idx % 9;
  long t  = idx / 9;
  long ci = t % Ci;
  long co = t / Ci;
  wt[(kk * Co + co) * Ci + ci] = (_Float16)w[idx];
}

__global__ void k_zero16(_Float16* __restrict__ z, int n) {
  int i = blockIdx.x * blockDim.x + threadIdx.x;
  if (i < n) z[i] = (_Float16)0.0f;
}

// ---------------------------------------------------------------------------
// Flatten one pyramid level: xe = src (f32 stream), xa16 = f16(src),
// q16 = f16(src + pos + level_embed).
// ---------------------------------------------------------------------------
__global__ void k_flatten(const float* __restrict__ src,
                          const float* __restrict__ pos,
                          const float* __restrict__ lvl_emb,
                          float* __restrict__ xe,
                          _Float16* __restrict__ xa16,
                          _Float16* __restrict__ q16,
                          int HW, int base) {
  long idx = (long)blockIdx.x * blockDim.x + threadIdx.x;
  long total = (long)BATCH * HW * D_MODEL;
  if (idx >= total) return;
  int  c = (int)(idx % D_MODEL);
  long t = idx / D_MODEL;
  long i = t % HW;
  long b = t / HW;
  long src_i = ((long)b * D_MODEL + c) * HW + i;   // NCHW
  float s = src[src_i];
  float p = pos[src_i];
  long row = b * (long)LV_TOT + base + i;
  xe[row * D_MODEL + c]   = s;
  xa16[row * D_MODEL + c] = (_Float16)s;
  q16[row * D_MODEL + c]  = (_Float16)(s + p + lvl_emb[c]);
}

// Decoder query: q = tgt + query_embed = 2*query_embed (tgt init = query_embed)
__global__ void k_dec_q(const float* __restrict__ qe, _Float16* __restrict__ q16,
                        long n) {
  long i = (long)blockIdx.x * blockDim.x + threadIdx.x;
  if (i < n) q16[i] = (_Float16)(2.0f * qe[i]);
}

// ---------------------------------------------------------------------------
// Tiled WMMA GEMM: C[M,N] = epilogue(A[M,K] @ Wt[N,K]^T + bias [+ res]).
// Block = 256 thr (8 waves), block tile 64x64, each wave 2 m-subtiles.
// M,N multiples of 64; K multiple of 32. EXEC all-ones throughout (ISA 7.12).
// LDS staging uses the gfx1250 async global->LDS copy (ASYNCcnt): each thread
// moves one 16B segment per matrix without a VGPR round-trip, then waits with
// s_wait_asynccnt before the block barrier.
// ---------------------------------------------------------------------------
template <bool RELU, bool ADDRES, bool OUTF16>
__global__ __launch_bounds__(256) void k_gemm(
    const _Float16* __restrict__ A, const _Float16* __restrict__ Wt,
    const float* __restrict__ bias, const float* __restrict__ res,
    float* __restrict__ Cf, _Float16* __restrict__ Ch, int M, int N, int K) {
  __shared__ _Float16 As[64 * 40];   // pad 32->40 halfs to dodge bank conflicts
  __shared__ _Float16 Bs[64 * 40];
  const int tid  = threadIdx.x;
  const int wid  = tid >> 5;
  const int lane = tid & 31;
  const int m0 = blockIdx.y * 64;
  const int n0 = blockIdx.x * 64;
  const int tn  = wid & 3;           // n subtile 0..3
  const int tmA = (wid >> 2) * 2;    // m subtiles {tmA, tmA+1}
  v8f acc0 = {}; v8f acc1 = {};
  const int lr = tid >> 2;           // coop-load row 0..63
  const int lc = (tid & 3) * 8;      // coop-load col 0,8,16,24
  const int fm  = lane & 15;
  const int kb  = (lane >> 4) * 8;   // A K-base per ISA layout
  const int kb2 = (lane >> 4) * 16;  // unused: B fragments use kb like A
  const _Float16* ga = A  + (size_t)(m0 + lr) * K + lc;
  const _Float16* gb = Wt + (size_t)(n0 + lr) * K + lc;
#if USE_ASYNC_LDS
  // Low 32 bits of a generic pointer into LDS are the LDS byte offset.
  unsigned la = (unsigned)(size_t)&As[lr * 40 + lc];
  unsigned lb = (unsigned)(size_t)&Bs[lr * 40 + lc];
#endif
  for (int k0 = 0; k0 < K; k0 += 32) {
#if USE_ASYNC_LDS
    unsigned long long pa = (unsigned long long)(ga + k0);
    unsigned long long pb = (unsigned long long)(gb + k0);
    asm volatile("global_load_async_to_lds_b128 %0, %1, off"
                 :: "v"(la), "v"(pa) : "memory");
    asm volatile("global_load_async_to_lds_b128 %0, %1, off"
                 :: "v"(lb), "v"(pb) : "memory");
    if (k0 + 32 < K) {                      // -> global_prefetch_b8
      __builtin_prefetch(ga + k0 + 32, 0, 1);
      __builtin_prefetch(gb + k0 + 32, 0, 1);
    }
    asm volatile("s_wait_asynccnt 0x0" ::: "memory");
#else
    v8h av = *(const v8h*)(ga + k0);
    v8h bv = *(const v8h*)(gb + k0);
    if (k0 + 32 < K) {
      __builtin_prefetch(ga + k0 + 32, 0, 1);
      __builtin_prefetch(gb + k0 + 32, 0, 1);
    }
    *(v8h*)&As[lr * 40 + lc] = av;
    *(v8h*)&Bs[lr * 40 + lc] = bv;
#endif
    __syncthreads();
    v8h a0lo = *(const v8h*)&As[(tmA * 16 + fm) * 40 + kb];
    v8h a0hi = *(const v8h*)&As[(tmA * 16 + fm) * 40 + kb + 16];
    v8h a1lo = *(const v8h*)&As[((tmA + 1) * 16 + fm) * 40 + kb];
    v8h a1hi = *(const v8h*)&As[((tmA + 1) * 16 + fm) * 40 + kb + 16];
    v8h blo  = *(const v8h*)&Bs[(tn * 16 + fm) * 40 + kb];
    v8h bhi  = *(const v8h*)&Bs[(tn * 16 + fm) * 40 + kb + 16];
    v16h af0 = cat8(a0lo, a0hi);
    v16h af1 = cat8(a1lo, a1hi);
    v16h bf  = cat8(blo, bhi);
    acc0 = __builtin_amdgcn_wmma_f32_16x16x32_f16(false, af0, false, bf,
                                                  (short)0, acc0, false, false);
    acc1 = __builtin_amdgcn_wmma_f32_16x16x32_f16(false, af1, false, bf,
                                                  (short)0, acc1, false, false);
    __syncthreads();
  }
  const int n  = n0 + tn * 16 + fm;
  const int mo = (lane >> 4) * 8;
  float bval = bias[n];
#pragma unroll
  for (int v = 0; v < 8; ++v) {
    {
      size_t idx = (size_t)(m0 + tmA * 16 + mo + v) * N + n;
      float r = acc0[v] + bval;
      if (ADDRES) r += res[idx];
      if (RELU)   r = fmaxf(r, 0.0f);
      if (OUTF16) Ch[idx] = (_Float16)r; else Cf[idx] = r;
    }
    {
      size_t idx = (size_t)(m0 + (tmA + 1) * 16 + mo + v) * N + n;
      float r = acc1[v] + bval;
      if (ADDRES) r += res[idx];
      if (RELU)   r = fmaxf(r, 0.0f);
      if (OUTF16) Ch[idx] = (_Float16)r; else Cf[idx] = r;
    }
  }
}

// ---------------------------------------------------------------------------
// Fused attention-weight softmax + multiscale bilinear sampling.
// One block per token row, one wave per head, lane = head-dim channel.
// valid_ratios == 1 (all masks false in this model).
// ---------------------------------------------------------------------------
template <bool DEC>
__global__ __launch_bounds__(256) void k_msda(
    const float* __restrict__ off, const float* __restrict__ aw_in,
    const float* __restrict__ value, const float* __restrict__ refpts,
    _Float16* __restrict__ out16, int Lrows) {
  const int Wl[4] = {64, 32, 16, 8};
  const int Hl[4] = {64, 32, 16, 8};
  const int st[4] = {0, 4096, 5120, 5376};
  long row  = blockIdx.x;
  int  h    = threadIdx.x >> 5;
  int  lane = threadIdx.x & 31;
  int  b    = (int)(row / Lrows);
  long q    = row % Lrows;
  float rx, ry;
  if (DEC) {
    rx = refpts[((long)b * Lrows + q) * 2 + 0];
    ry = refpts[((long)b * Lrows + q) * 2 + 1];
  } else {
    int lp = (q < 4096) ? 0 : (q < 5120) ? 1 : (q < 5376) ? 2 : 3;
    long loc = q - st[lp];
    int ww = Wl[lp];
    int iy = (int)(loc / ww), ix = (int)(loc % ww);
    rx = (ix + 0.5f) / (float)ww;
    ry = (iy + 0.5f) / (float)Hl[lp];
  }
  float a[16];
  const float* awp = aw_in + row * 128 + h * 16;
  float mx = -1e30f;
#pragma unroll
  for (int i = 0; i < 16; ++i) { a[i] = awp[i]; mx = fmaxf(mx, a[i]); }
  float se = 0.0f;
#pragma unroll
  for (int i = 0; i < 16; ++i) { a[i] = __expf(a[i] - mx); se += a[i]; }
  float inv = 1.0f / se;
  const float* offp = off + row * 256 + h * 32;
  float acc = 0.0f;
#pragma unroll
  for (int l = 0; l < 4; ++l) {
    int Wc = Wl[l], Hc = Hl[l];
    const float* vb = value + ((long)b * LV_TOT + st[l]) * D_MODEL + h * HDIM + lane;
#pragma unroll
    for (int pI = 0; pI < 4; ++pI) {
      int s = l * 4 + pI;
      float lx = rx + offp[s * 2 + 0] / (float)Wc;
      float ly = ry + offp[s * 2 + 1] / (float)Hc;
      float fx = lx * Wc - 0.5f, fy = ly * Hc - 0.5f;
      float x0f = floorf(fx), y0f = floorf(fy);
      int x0 = (int)x0f, y0 = (int)y0f;
      float wx = fx - x0f, wy = fy - y0f;
      float wgt = a[s] * inv;
      float smp = 0.0f;
#pragma unroll
      for (int dy = 0; dy < 2; ++dy)
#pragma unroll
        for (int dx = 0; dx < 2; ++dx) {
          int xi = x0 + dx, yi = y0 + dy;
          if (xi >= 0 && xi < Wc && yi >= 0 && yi < Hc) {
            float wq = (dy ? wy : 1.0f - wy) * (dx ? wx : 1.0f - wx);
            smp += wq * vb[((long)yi * Wc + xi) * D_MODEL];
          }
        }
      acc += wgt * smp;
    }
  }
  out16[row * D_MODEL + h * HDIM + lane] = (_Float16)acc;
}

// ---------------------------------------------------------------------------
// LayerNorm over D=256; one block per row; writes f32 (+ optional f16 copy).
// ---------------------------------------------------------------------------
__global__ __launch_bounds__(256) void k_ln(
    const float* __restrict__ in, const float* __restrict__ g,
    const float* __restrict__ bt, float* __restrict__ out,
    _Float16* __restrict__ out16) {
  __shared__ float red[256];
  long row = blockIdx.x;
  int  t   = threadIdx.x;
  float v = in[row * D_MODEL + t];
  red[t] = v;
  __syncthreads();
  for (int s = 128; s > 0; s >>= 1) { if (t < s) red[t] += red[t + s]; __syncthreads(); }
  float mean = red[0] * (1.0f / 256.0f);
  __syncthreads();
  float d = v - mean;
  red[t] = d * d;
  __syncthreads();
  for (int s = 128; s > 0; s >>= 1) { if (t < s) red[t] += red[t + s]; __syncthreads(); }
  float var = red[0] * (1.0f / 256.0f);
  float r = d * rsqrtf(var + 1e-5f) * g[t] + bt[t];
  out[row * D_MODEL + t] = r;
  if (out16) out16[row * D_MODEL + t] = (_Float16)r;
}

// ---------------------------------------------------------------------------
// Channels-last f16 packers (conv input staging).
// ---------------------------------------------------------------------------
// NCHW f32 -> (B, HW, Ctot) f16 at channel offset coff
__global__ void k_pack_nchw(const float* __restrict__ in, _Float16* __restrict__ ocl,
                            int C, long HW, int Ctot, int coff) {
  long idx = (long)blockIdx.x * blockDim.x + threadIdx.x;
  long total = (long)BATCH * C * HW;
  if (idx >= total) return;
  long p = idx % HW;
  long t = idx / HW;
  long c = t % C;
  long b = t / C;
  ocl[((long)b * HW + p) * Ctot + coff + c] = (_Float16)in[idx];
}

// (B, HW, C) f32 rows -> (B, HW, Ctot) f16 at channel offset coff
__global__ void k_pack_rows(const float* __restrict__ in, _Float16* __restrict__ ocl,
                            int C, long HW, int Ctot, int coff) {
  long idx = (long)blockIdx.x * blockDim.x + threadIdx.x;
  long total = (long)BATCH * HW * C;
  if (idx >= total) return;
  int  c = (int)(idx % C);
  long r = idx / C;                  // b*HW + p
  ocl[r * Ctot + coff + c] = (_Float16)in[idx];
}

// Bilinear x2 upsample (align_corners=True) NCHW f32 -> channels-last f16
__global__ void k_upsample_pack(const float* __restrict__ in, _Float16* __restrict__ ocl,
                                int C, int H, int W, int Ctot, int coff) {
  long idx = (long)blockIdx.x * blockDim.x + threadIdx.x;
  long HO = 2L * H, WO = 2L * W;
  long total = (long)BATCH * C * HO * WO;
  if (idx >= total) return;
  long x = idx % WO; long t = idx / WO;
  long y = t % HO;  t /= HO;
  long c = t % C;   long b = t / C;
  float fy = (float)y * (float)(H - 1) / (float)(HO - 1);
  float fx = (float)x * (float)(W - 1) / (float)(WO - 1);
  int y0 = (int)floorf(fy), x0 = (int)floorf(fx);
  int y1 = min(y0 + 1, H - 1), x1 = min(x0 + 1, W - 1);
  float wy = fy - y0, wx = fx - x0;
  const float* ip = in + ((b * C + c) * (long)H) * W;
  float v00 = ip[(long)y0 * W + x0], v01 = ip[(long)y0 * W + x1];
  float v10 = ip[(long)y1 * W + x0], v11 = ip[(long)y1 * W + x1];
  float r = v00 * (1 - wy) * (1 - wx) + v01 * (1 - wy) * wx +
            v10 * wy * (1 - wx) + v11 * wy * wx;
  ocl[((long)b * HO * WO + y * WO + x) * Ctot + coff + c] = (_Float16)r;
}

// ---------------------------------------------------------------------------
// Implicit-GEMM 3x3 conv (pad 1) + bias + ReLU via WMMA.
// Input is channels-last f16 (B, HW, Ci); one wave = 16 pixels x 16 Cout.
// Out-of-image taps read a zeroed scratch row via pointer select (no EXEC
// divergence, loads stay b128).
// ---------------------------------------------------------------------------
__global__ __launch_bounds__(256) void k_conv(
    const _Float16* __restrict__ cin, const _Float16* __restrict__ zbuf,
    const _Float16* __restrict__ Wt,  // [9][Co][Ci]
    const float* __restrict__ bias, float* __restrict__ out,  // NCHW f32
    int H, int W, int Ci, int Co) {
  const int lane = threadIdx.x & 31;
  const int wid  = threadIdx.x >> 5;
  const long gw  = (long)blockIdx.x * 8 + wid;
  const int tilesN = Co >> 4;
  const int  tn = (int)(gw % tilesN);
  const long tm = gw / tilesN;
  const int fm  = lane & 15;
  const int kb  = (lane >> 4) * 8;
  const int kb2 = (lane >> 4) * 16;
  const long HW = (long)H * W;
  const long m  = tm * 16 + fm;          // this lane's pixel row (A layout)
  const int  b  = (int)(m / HW);
  const long p  = m % HW;
  const int  y  = (int)(p / W), x = (int)(p % W);
  const int  n  = tn * 16 + fm;
  v8f acc = {};
#pragma unroll
  for (int kk = 0; kk < 9; ++kk) {
    int ky = kk / 3, kx = kk % 3;
    int yy = y + ky - 1, xx = x + kx - 1;
    bool valid = (yy >= 0) && (yy < H) && (xx >= 0) && (xx < W);
    const _Float16* abase = valid ? cin + ((long)b * HW + (long)yy * W + xx) * Ci
                                  : zbuf;
    const _Float16* wbase = Wt + ((long)kk * Co + n) * Ci;
    for (int cb = 0; cb < Ci; cb += 32) {
      v16h af = cat8(*(const v8h*)(abase + cb + kb),
                     *(const v8h*)(abase + cb + kb + 16));
      v16h bf = cat8(*(const v8h*)(wbase + cb + kb),
                     *(const v8h*)(wbase + cb + kb + 16));
      acc = __builtin_amdgcn_wmma_f32_16x16x32_f16(false, af, false, bf,
                                                   (short)0, acc, false, false);
    }
  }
  float bv = bias[n];
#pragma unroll
  for (int v = 0; v < 8; ++v) {
    long mm = tm * 16 + (lane >> 4) * 8 + v;
    int  bb = (int)(mm / HW);
    long pp = mm % HW;
    out[((long)bb * Co + n) * HW + pp] = fmaxf(acc[v] + bv, 0.0f);
  }
}

// ---------------------------------------------------------------------------
// Final conv (192->256) fused with fc1 (256->1) + sigmoid: one block per
// 16-pixel tile; 8 waves x 2 Cout tiles cover all 256 channels; channel
// reduction via ds_add_f32. Avoids materializing the 268MB logits tensor.
// ---------------------------------------------------------------------------
__global__ __launch_bounds__(256) void k_conv_fc(
    const _Float16* __restrict__ cin, const _Float16* __restrict__ zbuf,
    const _Float16* __restrict__ Wt, const float* __restrict__ bias,
    const float* __restrict__ fcw, const float* __restrict__ fcb,
    float* __restrict__ preds, int H, int W, int Ci) {
  const int Co = 256;
  __shared__ float ssum[16];
  if (threadIdx.x < 16) ssum[threadIdx.x] = 0.0f;
  __syncthreads();
  const int lane = threadIdx.x & 31;
  const int wid  = threadIdx.x >> 5;
  const long tm  = blockIdx.x;
  const int fm   = lane & 15;
  const int kb   = (lane >> 4) * 8;
  const int kb2  = (lane >> 4) * 16;
  const long HW  = (long)H * W;
  const long m   = tm * 16 + fm;
  const int  b   = (int)(m / HW);
  const long p   = m % HW;
  const int  y   = (int)(p / W), x = (int)(p % W);
  const int  tn0 = wid * 2;
  v8f acc0 = {}; v8f acc1 = {};
#pragma unroll
  for (int kk = 0; kk < 9; ++kk) {
    int ky = kk / 3, kx = kk % 3;
    int yy = y + ky - 1, xx = x + kx - 1;
    bool valid = (yy >= 0) && (yy < H) && (xx >= 0) && (xx < W);
    const _Float16* abase = valid ? cin + ((long)b * HW + (long)yy * W + xx) * Ci
                                  : zbuf;
    const _Float16* wb0 = Wt + ((long)kk * Co + tn0 * 16 + fm) * Ci;
    const _Float16* wb1 = Wt + ((long)kk * Co + (tn0 + 1) * 16 + fm) * Ci;
    for (int cb = 0; cb < Ci; cb += 32) {
      v16h af = cat8(*(const v8h*)(abase + cb + kb),
                     *(const v8h*)(abase + cb + kb + 16));
      v16h bf0 = cat8(*(const v8h*)(wb0 + cb + kb), *(const v8h*)(wb0 + cb + kb + 16));
      v16h bf1 = cat8(*(const v8h*)(wb1 + cb + kb), *(const v8h*)(wb1 + cb + kb + 16));
      acc0 = __builtin_amdgcn_wmma_f32_16x16x32_f16(false, af, false, bf0,
                                                    (short)0, acc0, false, false);
      acc1 = __builtin_amdgcn_wmma_f32_16x16x32_f16(false, af, false, bf1,
                                                    (short)0, acc1, false, false);
    }
  }
  const int mo = (lane >> 4) * 8;
#pragma unroll
  for (int v = 0; v < 8; ++v) {
    int co0 = tn0 * 16 + fm;
    int co1 = (tn0 + 1) * 16 + fm;
    float r0 = fmaxf(acc0[v] + bias[co0], 0.0f) * fcw[co0];
    float r1 = fmaxf(acc1[v] + bias[co1], 0.0f) * fcw[co1];
    atomicAdd(&ssum[mo + v], r0 + r1);
  }
  __syncthreads();
  if (threadIdx.x < 16) {
    long mm = tm * 16 + threadIdx.x;   // linear over B*H*W == preds layout
    float s = ssum[threadIdx.x] + fcb[0];
    preds[mm] = 1.0f / (1.0f + __expf(-s));
  }
}

// ---------------------------------------------------------------------------
// Host-side launch orchestration.
// ---------------------------------------------------------------------------
static inline size_t al256(size_t x) { return (x + 255) & ~(size_t)255; }

static void gemm(hipStream_t s, const _Float16* A, const _Float16* Wt,
                 const float* bias, const float* res, float* Cf, _Float16* Ch,
                 int M, int N, int K, bool relu, bool addres, bool outf16) {
  (void)relu;
  dim3 grid(N / 64, M / 64);
  if (outf16)      k_gemm<true,  false, true ><<<grid, 256, 0, s>>>(A, Wt, bias, res, Cf, Ch, M, N, K);
  else if (addres) k_gemm<false, true,  false><<<grid, 256, 0, s>>>(A, Wt, bias, res, Cf, Ch, M, N, K);
  else             k_gemm<false, false, false><<<grid, 256, 0, s>>>(A, Wt, bias, res, Cf, Ch, M, N, K);
}

extern "C" void kernel_launch(void* const* d_in, const int* in_sizes, int n_in,
                              void* d_out, int out_size, void* d_ws, size_t ws_size,
                              hipStream_t stream) {
  (void)in_sizes; (void)n_in; (void)out_size; (void)ws_size;
  // Input order: top-level dict insertion order; `params` pytree leaves in
  // sorted-key order (jax convention).
  const float* src[4]  = {(const float*)d_in[0], (const float*)d_in[3],
                          (const float*)d_in[6], (const float*)d_in[9]};
  const float* pos[4]  = {(const float*)d_in[1], (const float*)d_in[4],
                          (const float*)d_in[7], (const float*)d_in[10]};
  const float* query_embed = (const float*)d_in[12];
  const float* ref_pts     = (const float*)d_in[13];
  const float* layer1      = (const float*)d_in[14];
  const float* layer0      = (const float*)d_in[15];
  const float* x_orig      = (const float*)d_in[16];
  const float* c2_b  = (const float*)d_in[17];
  const float* c2_w  = (const float*)d_in[18];
  const float* cu0_b = (const float*)d_in[19];
  const float* cu0_w = (const float*)d_in[20];
  const float* cu1_b = (const float*)d_in[21];
  const float* cu1_w = (const float*)d_in[22];
  // dec: 23 aw_b 24 aw_w 25 o_b 26 o_w 27 off_b 28 off_w 29 v_b 30 v_w
  //      31 l1b 32 l1w 33 l2b 34 l2w 35 n1b 36 n1g 37 n2b 38 n2g
  // enc: 39..54 same order
  const float* d_aw_b  = (const float*)d_in[23]; const float* d_aw_w  = (const float*)d_in[24];
  const float* d_o_b   = (const float*)d_in[25]; const float* d_o_w   = (const float*)d_in[26];
  const float* d_off_b = (const float*)d_in[27]; const float* d_off_w = (const float*)d_in[28];
  const float* d_v_b   = (const float*)d_in[29]; const float* d_v_w   = (const float*)d_in[30];
  const float* d_l1b = (const float*)d_in[31]; const float* d_l1w = (const float*)d_in[32];
  const float* d_l2b = (const float*)d_in[33]; const float* d_l2w = (const float*)d_in[34];
  const float* d_n1b = (const float*)d_in[35]; const float* d_n1g = (const float*)d_in[36];
  const float* d_n2b = (const float*)d_in[37]; const float* d_n2g = (const float*)d_in[38];
  const float* e_aw_b  = (const float*)d_in[39]; const float* e_aw_w  = (const float*)d_in[40];
  const float* e_o_b   = (const float*)d_in[41]; const float* e_o_w   = (const float*)d_in[42];
  const float* e_off_b = (const float*)d_in[43]; const float* e_off_w = (const float*)d_in[44];
  const float* e_v_b   = (const float*)d_in[45]; const float* e_v_w   = (const float*)d_in[46];
  const float* e_l1b = (const float*)d_in[47]; const float* e_l1w = (const float*)d_in[48];
  const float* e_l2b = (const float*)d_in[49]; const float* e_l2w = (const float*)d_in[50];
  const float* e_n1b = (const float*)d_in[51]; const float* e_n1g = (const float*)d_in[52];
  const float* e_n2b = (const float*)d_in[53]; const float* e_n2g = (const float*)d_in[54];
  const float* fc1_b = (const float*)d_in[55];
  const float* fc1_w = (const float*)d_in[56];
  const float* lvl_emb = (const float*)d_in[57];
  float* preds = (float*)d_out;

  // ---- workspace bump allocation ----
  char* ws = (char*)d_ws;
  size_t cur = 0;
  auto alloc = [&](size_t bytes) { size_t o = cur; cur += al256(bytes); return ws + o; };
  // f16 transposed weights
  _Float16* wt_ev  = (_Float16*)alloc(65536 * 2);
  _Float16* wt_eo  = (_Float16*)alloc(65536 * 2);
  _Float16* wt_eow = (_Float16*)alloc(65536 * 2);
  _Float16* wt_ea  = (_Float16*)alloc(32768 * 2);
  _Float16* wt_e1  = (_Float16*)alloc(262144 * 2);
  _Float16* wt_e2  = (_Float16*)alloc(262144 * 2);
  _Float16* wt_dv  = (_Float16*)alloc(65536 * 2);
  _Float16* wt_do  = (_Float16*)alloc(65536 * 2);
  _Float16* wt_dow = (_Float16*)alloc(65536 * 2);
  _Float16* wt_da  = (_Float16*)alloc(32768 * 2);
  _Float16* wt_d1  = (_Float16*)alloc(262144 * 2);
  _Float16* wt_d2  = (_Float16*)alloc(262144 * 2);
  _Float16* wt_cu1 = (_Float16*)alloc((size_t)9 * 256 * 512 * 2);
  _Float16* wt_cu0 = (_Float16*)alloc((size_t)9 * 128 * 320 * 2);
  _Float16* wt_c2  = (_Float16*)alloc((size_t)9 * 256 * 192 * 2);
  _Float16* zbuf   = (_Float16*)alloc(1024 * 2);
  // activations
  float*    xe    = (float*)   alloc((size_t)RE * 256 * 4);
  _Float16* xa16  = (_Float16*)alloc((size_t)RE * 256 * 2);
  _Float16* q16   = (_Float16*)alloc((size_t)RE * 256 * 2);  // also attn16
  _Float16* h16   = (_Float16*)alloc((size_t)RE * 1024 * 2);
  _Float16* mem16 = (_Float16*)alloc((size_t)RE * 256 * 2);
  float*    value = (float*)   alloc((size_t)RE * 256 * 4);
  float*    offb  = (float*)   alloc((size_t)RE * 256 * 4);
  float*    awb   = (float*)   alloc((size_t)RE * 128 * 4);
  float*    t0    = (float*)   alloc((size_t)RE * 256 * 4);
  float*    tgt   = (float*)   alloc((size_t)RD * 256 * 4);  // becomes hs
  // head (channels-last f16 conv inputs + NCHW f32 conv outputs)
  _Float16* cin1 = (_Float16*)alloc((size_t)BATCH * 4096  * 512 * 2);
  float*    y1   = (float*)   alloc((size_t)BATCH * 256 * 64 * 64 * 4);
  _Float16* cin0 = (_Float16*)alloc((size_t)BATCH * 16384 * 320 * 2);
  float*    y0   = (float*)   alloc((size_t)BATCH * 128 * 128 * 128 * 4);
  _Float16* cin2 = (_Float16*)alloc((size_t)BATCH * 65536 * 192 * 2);

  auto WT = [&](const float* w, _Float16* wt, int K, int N) {
    long tot = (long)K * N;
    k_wt<<<(tot + 255) / 256, 256, 0, stream>>>(w, wt, K, N);
  };
  WT(e_v_w, wt_ev, 256, 256);    WT(e_o_w, wt_eo, 256, 256);
  WT(e_off_w, wt_eow, 256, 256); WT(e_aw_w, wt_ea, 256, 128);
  WT(e_l1w, wt_e1, 256, 1024);   WT(e_l2w, wt_e2, 1024, 256);
  WT(d_v_w, wt_dv, 256, 256);    WT(d_o_w, wt_do, 256, 256);
  WT(d_off_w, wt_dow, 256, 256); WT(d_aw_w, wt_da, 256, 128);
  WT(d_l1w, wt_d1, 256, 1024);   WT(d_l2w, wt_d2, 1024, 256);
  k_cwt<<<((long)256 * 512 * 9 + 255) / 256, 256, 0, stream>>>(cu1_w, wt_cu1, 256, 512);
  k_cwt<<<((long)128 * 320 * 9 + 255) / 256, 256, 0, stream>>>(cu0_w, wt_cu0, 128, 320);
  k_cwt<<<((long)256 * 192 * 9 + 255) / 256, 256, 0, stream>>>(c2_w, wt_c2, 256, 192);
  k_zero16<<<4, 256, 0, stream>>>(zbuf, 1024);

  // ---- flatten pyramid ----
  const int HWs[4]  = {4096, 1024, 256, 64};
  const int base[4] = {0, 4096, 5120, 5376};
  for (int l = 0; l < 4; ++l) {
    long tot = (long)BATCH * HWs[l] * 256;
    k_flatten<<<(tot + 255) / 256, 256, 0, stream>>>(
        src[l], pos[l], lvl_emb + l * 256, xe, xa16, q16, HWs[l], base[l]);
  }

  // ---- encoder layer ----
  gemm(stream, xa16, wt_ev,  e_v_b,   nullptr, value, nullptr, RE, 256, 256, false, false, false);
  gemm(stream, q16,  wt_eow, e_off_b, nullptr, offb,  nullptr, RE, 256, 256, false, false, false);
  gemm(stream, q16,  wt_ea,  e_aw_b,  nullptr, awb,   nullptr, RE, 128, 256, false, false, false);
  k_msda<false><<<RE, 256, 0, stream>>>(offb, awb, value, nullptr, q16, LV_TOT);
  gemm(stream, q16, wt_eo, e_o_b, xe, t0, nullptr, RE, 256, 256, false, true, false);
  k_ln<<<RE, 256, 0, stream>>>(t0, e_n1g, e_n1b, xe, xa16);
  gemm(stream, xa16, wt_e1, e_l1b, nullptr, nullptr, h16, RE, 1024, 256, true, false, true);
  gemm(stream, h16,  wt_e2, e_l2b, xe, t0, nullptr, RE, 256, 1024, false, true, false);
  k_ln<<<RE, 256, 0, stream>>>(t0, e_n2g, e_n2b, xe, mem16);  // xe := memory

  // ---- decoder layer (cross-attn into memory) ----
  gemm(stream, mem16, wt_dv, d_v_b, nullptr, value, nullptr, RE, 256, 256, false, false, false);
  k_dec_q<<<((long)RD * 256 + 255) / 256, 256, 0, stream>>>(query_embed, q16, (long)RD * 256);
  gemm(stream, q16, wt_dow, d_off_b, nullptr, offb, nullptr, RD, 256, 256, false, false, false);
  gemm(stream, q16, wt_da,  d_aw_b,  nullptr, awb,  nullptr, RD, 128, 256, false, false, false);
  k_msda<true><<<RD, 256, 0, stream>>>(offb, awb, value, ref_pts, q16, LQ);
  gemm(stream, q16, wt_do, d_o_b, query_embed, t0, nullptr, RD, 256, 256, false, true, false);
  k_ln<<<RD, 256, 0, stream>>>(t0, d_n1g, d_n1b, tgt, xa16);
  gemm(stream, xa16, wt_d1, d_l1b, nullptr, nullptr, h16, RD, 1024, 256, true, false, true);
  gemm(stream, h16,  wt_d2, d_l2b, tgt, t0, nullptr, RD, 256, 1024, false, true, false);
  k_ln<<<RD, 256, 0, stream>>>(t0, d_n2g, d_n2b, tgt, nullptr);  // tgt := hs

  // ---- corner head ----
  // pack conv1 input: channels 0..255 = hs rows, 256..511 = layer1 NCHW
  k_pack_rows<<<((long)BATCH * 4096 * 256 + 255) / 256, 256, 0, stream>>>(
      tgt, cin1, 256, 4096, 512, 0);
  k_pack_nchw<<<((long)BATCH * 256 * 4096 + 255) / 256, 256, 0, stream>>>(
      layer1, cin1, 256, 4096, 512, 256);
  {
    long waves = ((long)BATCH * 64 * 64 / 16) * (256 / 16);
    k_conv<<<waves / 8, 256, 0, stream>>>(cin1, zbuf, wt_cu1, cu1_b, y1,
                                          64, 64, 512, 256);
  }
  // pack conv0 input: 0..255 = upsample2(y1), 256..319 = layer0
  k_upsample_pack<<<((long)BATCH * 256 * 16384 + 255) / 256, 256, 0, stream>>>(
      y1, cin0, 256, 64, 64, 320, 0);
  k_pack_nchw<<<((long)BATCH * 64 * 16384 + 255) / 256, 256, 0, stream>>>(
      layer0, cin0, 64, 16384, 320, 256);
  {
    long waves = ((long)BATCH * 128 * 128 / 16) * (128 / 16);
    k_conv<<<waves / 8, 256, 0, stream>>>(cin0, zbuf, wt_cu0, cu0_b, y0,
                                          128, 128, 320, 128);
  }
  // pack conv2 input: 0..127 = upsample2(y0), 128..191 = x_original
  k_upsample_pack<<<((long)BATCH * 128 * 65536 + 255) / 256, 256, 0, stream>>>(
      y0, cin2, 128, 128, 128, 192, 0);
  k_pack_nchw<<<((long)BATCH * 64 * 65536 + 255) / 256, 256, 0, stream>>>(
      x_orig, cin2, 64, 65536, 192, 128);
  {
    long pix_tiles = (long)BATCH * 256 * 256 / 16;
    k_conv_fc<<<pix_tiles, 256, 0, stream>>>(cin2, zbuf, wt_c2, c2_b,
                                             fc1_w, fc1_b, preds, 256, 256, 192);
  }
}
